// ThreeBodySpringMassGraphModel_70205535420458
// MI455X (gfx1250) — matrix operand order repaired and checked by reading, not compiled, not run on hardware
//
#include <hip/hip_runtime.h>

// ---------------------------------------------------------------------------
// ThreeBodySpringMassGraphModel on MI455X (gfx1250), wave32 + WMMA f16.
// One 256-thread block (8 wave32) handles one (batch, net) pair.
// B-fragments (weights) are register-resident; activations stream via LDS.
// ---------------------------------------------------------------------------

typedef __attribute__((ext_vector_type(16))) _Float16 v16h;
typedef __attribute__((ext_vector_type(8)))  _Float16 v8h;
typedef __attribute__((ext_vector_type(8)))  float    v8f;

#define BDIM 256
constexpr int BATCH = 4096;
constexpr int PN    = 16;    // particles per graph
constexpr int SD    = 2;     // spatial dim
constexpr int H     = 128;   // hidden

// ---- dynamic LDS layout (byte offsets, all 16B aligned) ----
constexpr int OFF_WE1T = 0;                          // [128][32]  f16 (We1^T, K padded 16->32)
constexpr int OFF_WE2T = OFF_WE1T + 128 * 32  * 2;   // [128][128] f16 (We2^T)
constexpr int OFF_WN1T = OFF_WE2T + 128 * 128 * 2;   // [128][160] f16 (Wn1^T, K padded 134->160)
constexpr int OFF_WN2T = OFF_WN1T + 128 * 160 * 2;   // [16][128]  f16 (Wn2^T, N padded 2->16)
constexpr int OFF_EINP = OFF_WN2T + 16  * 128 * 2;   // [256][32]  f16 edge-MLP input (K padded)
constexpr int OFF_MSG  = OFF_EINP + 256 * 32  * 2;   // [256][128] f16 relu(h1) for all edges
constexpr int OFF_NIN  = OFF_MSG  + 256 * 128 * 2;   // [16][160]  f16 node-MLP input
constexpr int OFF_HN   = OFF_NIN  + 16 * 160 * 2;    // [16][128]  f16 relu(node hidden)
constexpr int OFF_NODE = OFF_HN   + 16 * 128 * 2;    // [16][8]    f16 node features
constexpr int OFF_BE1  = OFF_NODE + 16 * 8 * 2;      // [128] f32
constexpr int OFF_BE2  = OFF_BE1  + 128 * 4;         // [128] f32
constexpr int OFF_BN1  = OFF_BE2  + 128 * 4;         // [128] f32
constexpr int OFF_BN2  = OFF_BN1  + 128 * 4;         // [2]   f32 (+pad)
constexpr int SMEM_BYTES = OFF_BN2 + 16;             // ~179 KB (<= 320 KB WGP LDS)

// A-fragment (16x32 f16): lane holds row M=lane&15; K = kb..kb+15 contiguous
// (kb=(lane>=16)*16)  -> two aligned 16B LDS loads.
__device__ __forceinline__ v16h ldsA(const _Float16* p) {
    v8h lo = *(const v8h*)(p);
    v8h hi = *(const v8h*)(p + 8);
    v16h r;
#pragma unroll
    for (int i = 0; i < 8; ++i) { r[i] = lo[i]; r[i + 8] = hi[i]; }
    return r;
}
// B-fragment (32x16 f16) from N-major (transposed) weights: lane holds column
// N=lane&15, K = kb..kb+15 contiguous (kb=(lane>=16)*16) -> two 16B LDS loads.
__device__ __forceinline__ v16h ldsB(const _Float16* p) {
    v8h lo = *(const v8h*)(p);
    v8h hi = *(const v8h*)(p + 8);
    v16h r;
#pragma unroll
    for (int i = 0; i < 8; ++i) { r[i] = lo[i]; r[i + 8] = hi[i]; }
    return r;
}
__device__ __forceinline__ v8f wmma16(v16h a, v16h b, v8f c) {
    return __builtin_amdgcn_wmma_f32_16x16x32_f16(false, a, false, b,
                                                  (short)0, c, false, false);
}
__device__ __forceinline__ v8f bcast8(float x) {
    v8f r;
#pragma unroll
    for (int i = 0; i < 8; ++i) r[i] = x;
    return r;
}
// pack two f32 -> two f16 -> one dword LDS store (dst must be 4B aligned)
__device__ __forceinline__ void st2h(_Float16* dst, float a, float b) {
    union { _Float16 h[2]; unsigned u; } u2;
    u2.h[0] = (_Float16)a; u2.h[1] = (_Float16)b;
    *(unsigned*)dst = u2.u;
}

__global__ __launch_bounds__(BDIM)
void gnet_kernel(const float* __restrict__ q,   const float* __restrict__ p,
                 const float* __restrict__ dq,  const float* __restrict__ dp,
                 const float* __restrict__ mass,
                 const float* __restrict__ len, const float* __restrict__ kspr,
                 const float* __restrict__ qWe1, const float* __restrict__ qbe1,
                 const float* __restrict__ qWe2, const float* __restrict__ qbe2,
                 const float* __restrict__ qWn1, const float* __restrict__ qbn1,
                 const float* __restrict__ qWn2, const float* __restrict__ qbn2,
                 const float* __restrict__ pWe1, const float* __restrict__ pbe1,
                 const float* __restrict__ pWe2, const float* __restrict__ pbe2,
                 const float* __restrict__ pWn1, const float* __restrict__ pbn1,
                 const float* __restrict__ pWn2, const float* __restrict__ pbn2,
                 float* __restrict__ out)
{
    extern __shared__ __align__(16) char smem[];
    _Float16* We1T   = (_Float16*)(smem + OFF_WE1T);
    _Float16* We2T   = (_Float16*)(smem + OFF_WE2T);
    _Float16* Wn1T   = (_Float16*)(smem + OFF_WN1T);
    _Float16* Wn2T   = (_Float16*)(smem + OFF_WN2T);
    _Float16* einp   = (_Float16*)(smem + OFF_EINP);
    _Float16* msgbuf = (_Float16*)(smem + OFF_MSG);
    _Float16* nin    = (_Float16*)(smem + OFF_NIN);
    _Float16* hn     = (_Float16*)(smem + OFF_HN);
    _Float16* nodef  = (_Float16*)(smem + OFF_NODE);
    float*    be1s   = (float*)(smem + OFF_BE1);
    float*    be2s   = (float*)(smem + OFF_BE2);
    float*    bn1s   = (float*)(smem + OFF_BN1);
    float*    bn2s   = (float*)(smem + OFF_BN2);

    const int tid = threadIdx.x;
    const int bb  = blockIdx.x >> 1;   // batch element
    const int net = blockIdx.x & 1;    // 0 = q-net, 1 = p-net

    const float* X   = net ? p    : q;
    const float* DX  = net ? dp   : dq;
    const float* We1 = net ? pWe1 : qWe1;
    const float* be1 = net ? pbe1 : qbe1;
    const float* We2 = net ? pWe2 : qWe2;
    const float* be2 = net ? pbe2 : qbe2;
    const float* Wn1 = net ? pWn1 : qWn1;
    const float* bn1 = net ? pbn1 : qbn1;
    const float* Wn2 = net ? pWn2 : qWn2;
    const float* bn2 = net ? pbn2 : qbn2;

    // -------- stage weights (transposed, zero-padded, f16, paired stores) ---
    for (int idx = tid * 2; idx < 128 * 32; idx += BDIM * 2) {
        int n = idx >> 5, kk = idx & 31;                   // We1 is [16][128]
        st2h(&We1T[idx],
             (kk     < 16) ? We1[kk * H + n]       : 0.f,
             (kk + 1 < 16) ? We1[(kk + 1) * H + n] : 0.f);
    }
    for (int idx = tid * 2; idx < 128 * 128; idx += BDIM * 2) {
        int n = idx >> 7, kk = idx & 127;                  // We2 is [128][128]
        st2h(&We2T[idx], We2[kk * H + n], We2[(kk + 1) * H + n]);
    }
    for (int idx = tid * 2; idx < 128 * 160; idx += BDIM * 2) {
        int n = idx / 160, kk = idx % 160;                 // Wn1 is [134][128]
        st2h(&Wn1T[idx],
             (kk     < 134) ? Wn1[kk * H + n]       : 0.f,
             (kk + 1 < 134) ? Wn1[(kk + 1) * H + n] : 0.f);
    }
    for (int idx = tid * 2; idx < 16 * 128; idx += BDIM * 2) {
        int n = idx >> 7, kk = idx & 127;                  // Wn2 is [128][2]
        if (n < SD) st2h(&Wn2T[idx], Wn2[kk * SD + n], Wn2[(kk + 1) * SD + n]);
        else        st2h(&Wn2T[idx], 0.f, 0.f);
    }
    if (tid < 128) { be1s[tid] = be1[tid]; be2s[tid] = be2[tid]; bn1s[tid] = bn1[tid]; }
    if (tid < 2)   { bn2s[tid] = bn2[tid]; }
    if (tid < PN) {                                        // xf = [q0,dq0,m,q1,dq1,m]
        int g = bb * PN + tid;
        float q0 = X[g * 2 + 0], q1 = X[g * 2 + 1];
        float d0 = DX[g * 2 + 0], d1 = DX[g * 2 + 1];
        float mm = mass[g];
        _Float16* nd = nodef + tid * 8;
        nd[0] = (_Float16)q0; nd[1] = (_Float16)d0; nd[2] = (_Float16)mm;
        nd[3] = (_Float16)q1; nd[4] = (_Float16)d1; nd[5] = (_Float16)mm;
        nd[6] = (_Float16)0.f; nd[7] = (_Float16)0.f;
    }
    __syncthreads();

    // ------------- build edge-MLP inputs + node-MLP static columns ----------
    {   // edge e = tid: receiver r = e>>4, sender s = e&15
        int r = tid >> 4, s = tid & 15;
        float lv = len [bb * 256 + tid];
        float kv = kspr[bb * 256 + tid];
        _Float16* e = einp + tid * 32;
#pragma unroll
        for (int i = 0; i < 6; ++i) { e[i] = nodef[s * 8 + i]; e[6 + i] = nodef[r * 8 + i]; }
        e[12] = (_Float16)lv; e[13] = (_Float16)kv;
        e[14] = (_Float16)lv; e[15] = (_Float16)kv;
#pragma unroll
        for (int i = 16; i < 32; i += 2) st2h(&e[i], 0.f, 0.f);
    }
    for (int idx = tid; idx < 16 * 160; idx += BDIM) {     // nin cols 0..5 and pad
        int row = idx / 160, c = idx % 160;
        if (c < 6)         nin[idx] = nodef[row * 8 + c];
        else if (c >= 134) nin[idx] = (_Float16)0.f;       // cols 6..133 = agg (later)
    }
    __syncthreads();

    const int lane = tid & 31;
    const int wave = tid >> 5;
    const int mrow = lane & 15;            // A-fragment row
    const int ncol = lane & 15;            // B/C/D column index
    const int kbA  = (lane >> 4) * 16;     // A-frag K base
    const int kbB  = (lane >> 4) * 16;     // B-frag K base
    const int mhi  = (lane >> 4) * 8;      // C/D M offset

    // ============ edge layer 1: wave = 2 M-tiles, We1 frags resident ========
    {
        v16h bWe1[8];
        float be1v[8];
#pragma unroll
        for (int nt = 0; nt < 8; ++nt) {
            bWe1[nt] = ldsB(&We1T[(nt * 16 + ncol) * 32 + kbB]);
            be1v[nt] = be1s[nt * 16 + ncol];
        }
#pragma unroll
        for (int t = 0; t < 2; ++t) {
            const int r = wave * 2 + t;    // M-tile (== receiver) index
            v16h a1 = ldsA(&einp[(r * 16 + mrow) * 32 + kbA]);
#pragma unroll
            for (int nt = 0; nt < 8; ++nt) {
                v8f c = bcast8(be1v[nt]);
                c = wmma16(a1, bWe1[nt], c);
#pragma unroll
                for (int g = 0; g < 8; ++g)
                    msgbuf[(r * 16 + g + mhi) * 128 + nt * 16 + ncol] =
                        (_Float16)fmaxf(c[g], 0.f);
            }
        }
    }
    __syncthreads();

    // ====== edge layer 2 + segment-sum: wave = 8 M-tiles x 2 N-tiles ========
    {
        const int mg = wave & 1;           // M-tiles mg*8 .. mg*8+7
        const int ng = wave >> 1;          // N-tiles ng*2, ng*2+1
        v16h bWe2[2][4];
        float be2v[2];
#pragma unroll
        for (int j = 0; j < 2; ++j) {
            const int nt = ng * 2 + j;
            be2v[j] = be2s[nt * 16 + ncol];
#pragma unroll
            for (int kt = 0; kt < 4; ++kt)
                bWe2[j][kt] = ldsB(&We2T[(nt * 16 + ncol) * 128 + kt * 32 + kbB]);
        }
#pragma unroll
        for (int mi = 0; mi < 8; ++mi) {
            const int m = mg * 8 + mi;     // M-tile (== receiver) index
            v16h a[4];
#pragma unroll
            for (int kt = 0; kt < 4; ++kt)
                a[kt] = ldsA(&msgbuf[(m * 16 + mrow) * 128 + kt * 32 + kbA]);
#pragma unroll
            for (int j = 0; j < 2; ++j) {
                v8f c = bcast8(be2v[j]);
#pragma unroll
                for (int kt = 0; kt < 4; ++kt)
                    c = wmma16(a[kt], bWe2[j][kt], c);
                // agg[m][n] = sum over all 16 senders (rows of this M-tile)
                float part = 0.f;
#pragma unroll
                for (int g = 0; g < 8; ++g) part += c[g];
                float tot = part + __shfl_xor(part, 16, 32);
                if (lane < 16)
                    nin[m * 160 + 6 + (ng * 2 + j) * 16 + lane] = (_Float16)tot;
            }
        }
    }
    __syncthreads();

    // ===================== node MLP layer 1 (wave = N-tile) =================
    {
        v8f c = bcast8(bn1s[wave * 16 + ncol]);
#pragma unroll
        for (int kt = 0; kt < 5; ++kt) {   // K = 160 (134 padded)
            v16h a  = ldsA(&nin[mrow * 160 + kt * 32 + kbA]);
            v16h bf = ldsB(&Wn1T[(wave * 16 + ncol) * 160 + kt * 32 + kbB]);
            c = wmma16(a, bf, c);
        }
#pragma unroll
        for (int g = 0; g < 8; ++g)
            hn[(g + mhi) * 128 + wave * 16 + ncol] = (_Float16)fmaxf(c[g], 0.f);
    }
    __syncthreads();

    // ================= node MLP layer 2 (wave 0) + global store =============
    if (wave == 0) {
        v8f c = bcast8(0.f);
#pragma unroll
        for (int kt = 0; kt < 4; ++kt) {
            v16h a  = ldsA(&hn[mrow * 128 + kt * 32 + kbA]);
            v16h bf = ldsB(&Wn2T[ncol * 128 + kt * 32 + kbB]);
            c = wmma16(a, bf, c);
        }
        if (ncol < SD) {
            float bv = bn2s[ncol];
            float* op = out + (size_t)net * BATCH * PN * SD;
#pragma unroll
            for (int g = 0; g < 8; ++g)
                op[(bb * PN + (g + mhi)) * SD + ncol] = c[g] + bv;
        }
    }
}

extern "C" void kernel_launch(void* const* d_in, const int* in_sizes, int n_in,
                              void* d_out, int out_size, void* d_ws, size_t ws_size,
                              hipStream_t stream) {
    // setup_inputs order:
    // 0:q 1:p 2:dq 3:dp 4:m 5:t 6:dt 7:length 8:k
    // 9:qWe1 10:qbe1 11:qWe2 12:qbe2 13:qWn1 14:qbn1 15:qWn2 16:qbn2
    // 17:pWe1 18:pbe1 19:pWe2 20:pbe2 21:pWn1 22:pbn1 23:pWn2 24:pbn2
    gnet_kernel<<<dim3(2 * BATCH), dim3(BDIM), SMEM_BYTES, stream>>>(
        (const float*)d_in[0],  (const float*)d_in[1],
        (const float*)d_in[2],  (const float*)d_in[3],
        (const float*)d_in[4],
        (const float*)d_in[7],  (const float*)d_in[8],
        (const float*)d_in[9],  (const float*)d_in[10],
        (const float*)d_in[11], (const float*)d_in[12],
        (const float*)d_in[13], (const float*)d_in[14],
        (const float*)d_in[15], (const float*)d_in[16],
        (const float*)d_in[17], (const float*)d_in[18],
        (const float*)d_in[19], (const float*)d_in[20],
        (const float*)d_in[21], (const float*)d_in[22],
        (const float*)d_in[23], (const float*)d_in[24],
        (float*)d_out);
}
